// PatchEncoder_49976239456535
// MI455X (gfx1250) — matrix operand order repaired and checked by reading, not compiled, not run on hardware
//
#include <hip/hip_runtime.h>
#include <hip/hip_bf16.h>

// ---------- types for WMMA ----------
typedef __attribute__((ext_vector_type(16))) __bf16 v16bf;
typedef __attribute__((ext_vector_type(8)))  __bf16 v8bf;
typedef __attribute__((ext_vector_type(8)))  float  v8f;

#define BN_TOT 32768   // B*N rows
#define P_DIM  768
#define H_DIM  1024
#define NOUT   256     // Vd*D
#define DEPTH  4

__device__ __forceinline__ __bf16 f2bf(float f) {
  unsigned int u = __builtin_bit_cast(unsigned int, f);
  unsigned int r = u + 0x7FFFu + ((u >> 16) & 1u);   // round-to-nearest-even
  unsigned short hs = (unsigned short)(r >> 16);
  return __builtin_bit_cast(__bf16, hs);
}

__device__ __forceinline__ float gelu_exact(float x) {
  return 0.5f * x * (1.0f + erff(x * 0.7071067811865475f));
}

// ------------------------------------------------------------------
// fp32 -> bf16 bulk convert (grid-stride, 4 elems/thread/iter)
// ------------------------------------------------------------------
__global__ __launch_bounds__(256)
void cvt_bf16_kernel(const float* __restrict__ src, __bf16* __restrict__ dst,
                     int n4) {
  for (int i = blockIdx.x * 256 + threadIdx.x; i < n4;
       i += gridDim.x * 256) {
    float4 v = ((const float4*)src)[i];
    dst[i * 4 + 0] = f2bf(v.x);
    dst[i * 4 + 1] = f2bf(v.y);
    dst[i * 4 + 2] = f2bf(v.z);
    dst[i * 4 + 3] = f2bf(v.w);
  }
}

// ------------------------------------------------------------------
// Tiled bf16 GEMM: epi(A[M,K] @ W[K,N] + bias[N] [+ res])
// Macro tile 128x128, 256 threads = 8 waves (4 along M x 2 along N),
// each wave owns 32x64 = 2x4 WMMA 16x16 f32 accumulators.
// K stepped by 32 through LDS; A frags via 16B LDS loads, B frags via strided 2B LDS loads.
// EPI: 0 = bias, 1 = bias+gelu, 2 = bias+residual-add(res fp32)
// WRITEF: store fp32 to Outf;  WRITEB: store bf16 to Outb
// launch_bounds(256,1): 8 waves/block, full VGPR budget -> no spills
// ------------------------------------------------------------------
template <int EPI, bool WRITEF, bool WRITEB>
__global__ __launch_bounds__(256, 1)
void wmma_gemm_kernel(const __bf16* __restrict__ A,
                      const __bf16* __restrict__ W,
                      const float* __restrict__ bias,
                      const float* __restrict__ res,
                      float* __restrict__ Outf,
                      __bf16* __restrict__ Outb,
                      int K, int N) {
  __shared__ __bf16 ldsA[128 * 32];   // [m][k] row-major, 8 KB
  __shared__ __bf16 ldsB[32 * 128];   // [k][n] row-major, 8 KB

  const int tid  = threadIdx.x;
  const int lane = tid & 31;
  const int wave = tid >> 5;
  const int half = lane >> 4;
  const int l16  = lane & 15;
  const int waveM = wave >> 1;        // 0..3 -> 32-row slab
  const int waveN = wave & 1;         // 0..1 -> 64-col slab

  const int mBase = blockIdx.y * 128;
  const int nBase = blockIdx.x * 128;

  // staging coordinates (16-byte chunks)
  const int ar = tid >> 1;            // A row 0..127
  const int ac = (tid & 1) * 16;      // A col chunk {0,16} -> two v8bf
  const int bk = tid >> 4;            // B k-row 0..15 (x2 rows per thread)
  const int bn = (tid & 15) * 8;      // B n chunk

  v8f acc[2][4] = {};

  for (int kk = 0; kk < K; kk += 32) {
    // ---- stage: issue all 4 global 16B loads, then all LDS stores ----
    const __bf16* gpa = &A[(size_t)(mBase + ar) * K + kk + ac];
    const __bf16* gp0 = &W[(size_t)(kk + bk) * N + nBase + bn];
    const __bf16* gp1 = &W[(size_t)(kk + bk + 16) * N + nBase + bn];
    v8bf ta0 = *(const v8bf*)(gpa);
    v8bf ta1 = *(const v8bf*)(gpa + 8);
    v8bf tb0 = *(const v8bf*)(gp0);
    v8bf tb1 = *(const v8bf*)(gp1);
    *(v8bf*)&ldsA[ar * 32 + ac]         = ta0;
    *(v8bf*)&ldsA[ar * 32 + ac + 8]     = ta1;
    *(v8bf*)&ldsB[bk * 128 + bn]        = tb0;
    *(v8bf*)&ldsB[(bk + 16) * 128 + bn] = tb1;

    // prefetch next K tile while this one computes
    if (kk + 32 < K) {
      __builtin_prefetch(&A[(size_t)(mBase + ar) * K + kk + 32 + ac], 0, 1);
      __builtin_prefetch(&W[(size_t)(kk + 32 + bk) * N + nBase + bn], 0, 1);
    }
    __syncthreads();

    // A fragment: lane l16 = row M; K runs [half*8..+7] and [16+half*8..+7]
    v16bf afrag[2], bfrag[4];
    for (int mt = 0; mt < 2; ++mt) {
      const __bf16* rp = &ldsA[(waveM * 32 + mt * 16 + l16) * 32];
      v8bf lo = *(const v8bf*)(rp + half * 8);
      v8bf hi = *(const v8bf*)(rp + 16 + half * 8);
      for (int i = 0; i < 8; ++i) { afrag[mt][i] = lo[i]; afrag[mt][i + 8] = hi[i]; }
    }
    // B fragment: lane l16 = col N; K runs [half*8..+7] and [16+half*8..+7]
    for (int nt = 0; nt < 4; ++nt) {
      const __bf16* rp = &ldsB[waveN * 64 + nt * 16 + l16];
      for (int i = 0; i < 8; ++i) { bfrag[nt][i] = rp[(half * 8 + i) * 128]; bfrag[nt][i + 8] = rp[(16 + half * 8 + i) * 128]; }
    }

    for (int mt = 0; mt < 2; ++mt)
      for (int nt = 0; nt < 4; ++nt)
        acc[mt][nt] = __builtin_amdgcn_wmma_f32_16x16x32_bf16(
            false, afrag[mt], false, bfrag[nt],
            (short)0, acc[mt][nt], false, false);
    __syncthreads();
  }

  // Epilogue: C/D layout -> lane n = l16, VGPR v -> m = v + 8*half
  for (int mt = 0; mt < 2; ++mt) {
    for (int nt = 0; nt < 4; ++nt) {
      const int gn = nBase + waveN * 64 + nt * 16 + l16;
      const float bv = bias[gn];
      for (int v = 0; v < 8; ++v) {
        const int gm = mBase + waveM * 32 + mt * 16 + v + 8 * half;
        float rr = acc[mt][nt][v] + bv;
        if (EPI == 1) rr = gelu_exact(rr);
        if (EPI == 2) rr += res[(size_t)gm * N + gn];
        if (WRITEF) Outf[(size_t)gm * N + gn] = rr;
        if (WRITEB) Outb[(size_t)gm * N + gn] = f2bf(rr);
      }
    }
  }
}

// ------------------------------------------------------------------
// Row LayerNorm over H_DIM: x = (h-mu)*rsqrt(var+eps)*g + b  (bf16 out)
// ------------------------------------------------------------------
__global__ __launch_bounds__(256)
void ln_kernel(const float* __restrict__ h, const float* __restrict__ g,
               const float* __restrict__ b, __bf16* __restrict__ x) {
  __shared__ float s1[256], s2[256];
  const int row = blockIdx.x;
  const float* r = h + (size_t)row * H_DIM;
  float a = 0.f, aa = 0.f;
  for (int i = threadIdx.x; i < H_DIM; i += 256) { float v = r[i]; a += v; aa += v * v; }
  s1[threadIdx.x] = a; s2[threadIdx.x] = aa;
  __syncthreads();
  for (int off = 128; off > 0; off >>= 1) {
    if (threadIdx.x < off) {
      s1[threadIdx.x] += s1[threadIdx.x + off];
      s2[threadIdx.x] += s2[threadIdx.x + off];
    }
    __syncthreads();
  }
  const float mu  = s1[0] * (1.0f / H_DIM);
  const float var = s2[0] * (1.0f / H_DIM) - mu * mu;
  const float rs  = rsqrtf(var + 1e-5f);
  for (int i = threadIdx.x; i < H_DIM; i += 256)
    x[(size_t)row * H_DIM + i] = f2bf((r[i] - mu) * rs * g[i] + b[i]);
}

// ------------------------------------------------------------------
// Per row: normalize 64 D=4 vectors, Gram 4x4, Jacobi eigenvalues,
// token = [sqrt(w_desc), log1p|w|, exp(-w), e1..e4]
// ------------------------------------------------------------------
__global__ __launch_bounds__(256)
void finalize_kernel(const float* __restrict__ Mbuf, float* __restrict__ out) {
  const int row = blockIdx.x * 256 + threadIdx.x;
  if (row >= BN_TOT) return;
  const float* mp = Mbuf + (size_t)row * NOUT;

  float G00 = 0.f, G01 = 0.f, G02 = 0.f, G03 = 0.f;
  float G11 = 0.f, G12 = 0.f, G13 = 0.f, G22 = 0.f, G23 = 0.f, G33 = 0.f;
  for (int v = 0; v < 64; ++v) {
    float4 a = ((const float4*)mp)[v];
    float nrm = sqrtf(a.x * a.x + a.y * a.y + a.z * a.z + a.w * a.w);
    float inv = 1.0f / fmaxf(nrm, 1e-12f);
    float a0 = a.x * inv, a1 = a.y * inv, a2 = a.z * inv, a3 = a.w * inv;
    G00 += a0 * a0; G01 += a0 * a1; G02 += a0 * a2; G03 += a0 * a3;
    G11 += a1 * a1; G12 += a1 * a2; G13 += a1 * a3;
    G22 += a2 * a2; G23 += a2 * a3; G33 += a3 * a3;
  }

  float Am[4][4] = {{G00, G01, G02, G03},
                    {G01, G11, G12, G13},
                    {G02, G12, G22, G23},
                    {G03, G13, G23, G33}};

  // cyclic Jacobi (eigenvalues only), 6 sweeps
  for (int sweep = 0; sweep < 6; ++sweep) {
    for (int p = 0; p < 3; ++p) {
      for (int q = p + 1; q < 4; ++q) {
        float apq = Am[p][q];
        if (fabsf(apq) > 1e-24f) {
          float tau = (Am[q][q] - Am[p][p]) / (2.0f * apq);
          float t = (tau >= 0.f ? 1.f : -1.f) / (fabsf(tau) + sqrtf(1.f + tau * tau));
          float c = rsqrtf(1.f + t * t);
          float s = t * c;
          for (int k = 0; k < 4; ++k) {           // columns: A = A*J
            float akp = Am[k][p], akq = Am[k][q];
            Am[k][p] = c * akp - s * akq;
            Am[k][q] = s * akp + c * akq;
          }
          for (int k = 0; k < 4; ++k) {           // rows: A = J^T*A
            float apk = Am[p][k], aqk = Am[q][k];
            Am[p][k] = c * apk - s * aqk;
            Am[q][k] = s * apk + c * aqk;
          }
        }
      }
    }
  }

  float w0 = Am[0][0], w1 = Am[1][1], w2 = Am[2][2], w3 = Am[3][3];
  #define SWP(a, b) { if (a < b) { float _t = a; a = b; b = _t; } }
  SWP(w0, w1); SWP(w2, w3); SWP(w0, w2); SWP(w1, w3); SWP(w1, w2);
  #undef SWP

  const float e1 = w0 + w1 + w2 + w3;
  const float e2 = w0 * w1 + w0 * w2 + w0 * w3 + w1 * w2 + w1 * w3 + w2 * w3;
  const float e3 = w0 * w1 * w2 + w0 * w1 * w3 + w0 * w2 * w3 + w1 * w2 * w3;
  const float e4 = w0 * w1 * w2 * w3;

  float* o = out + (size_t)row * 16;
  o[0]  = sqrtf(fmaxf(w0, 1e-24f));
  o[1]  = sqrtf(fmaxf(w1, 1e-24f));
  o[2]  = sqrtf(fmaxf(w2, 1e-24f));
  o[3]  = sqrtf(fmaxf(w3, 1e-24f));
  o[4]  = log1pf(fabsf(w0));
  o[5]  = log1pf(fabsf(w1));
  o[6]  = log1pf(fabsf(w2));
  o[7]  = log1pf(fabsf(w3));
  o[8]  = expf(-w0);
  o[9]  = expf(-w1);
  o[10] = expf(-w2);
  o[11] = expf(-w3);
  o[12] = e1; o[13] = e2; o[14] = e3; o[15] = e4;
}

// ------------------------------------------------------------------
extern "C" void kernel_launch(void* const* d_in, const int* in_sizes, int n_in,
                              void* d_out, int out_size, void* d_ws, size_t ws_size,
                              hipStream_t stream) {
  (void)in_sizes; (void)n_in; (void)out_size; (void)ws_size;
  const float* patches   = (const float*)d_in[0];
  const float* enc_in_w  = (const float*)d_in[1];
  const float* enc_in_b  = (const float*)d_in[2];
  const float* blk_ln_g  = (const float*)d_in[3];
  const float* blk_ln_b  = (const float*)d_in[4];
  const float* blk_w1    = (const float*)d_in[5];
  const float* blk_b1    = (const float*)d_in[6];
  const float* blk_w2    = (const float*)d_in[7];
  const float* blk_b2    = (const float*)d_in[8];
  const float* enc_out_w = (const float*)d_in[9];
  const float* enc_out_b = (const float*)d_in[10];

  // -------- workspace layout (bytes) --------
  char* w = (char*)d_ws;
  float*  h   = (float*)w;                          w += (size_t)BN_TOT * H_DIM * 4;   // 128 MB
  float*  Mb  = (float*)w;                          w += (size_t)BN_TOT * NOUT * 4;    //  32 MB
  __bf16* xb  = (__bf16*)w;                         w += (size_t)BN_TOT * H_DIM * 2;   //  64 MB
  __bf16* tb  = (__bf16*)w;                         w += (size_t)BN_TOT * H_DIM * 2;   //  64 MB
  __bf16* hb  = (__bf16*)w;                         w += (size_t)BN_TOT * H_DIM * 2;   //  64 MB
  __bf16* pb  = (__bf16*)w;                         w += (size_t)BN_TOT * P_DIM * 2;   //  48 MB
  __bf16* wib = (__bf16*)w;                         w += (size_t)P_DIM * H_DIM * 2;
  __bf16* w1b = (__bf16*)w;                         w += (size_t)DEPTH * H_DIM * H_DIM * 2;
  __bf16* w2b = (__bf16*)w;                         w += (size_t)DEPTH * H_DIM * H_DIM * 2;
  __bf16* wob = (__bf16*)w;                         w += (size_t)H_DIM * NOUT * 2;

  const dim3 blk(256);
  const dim3 gH(H_DIM / 128, BN_TOT / 128);         // (8, 256)
  const dim3 gO(NOUT / 128,  BN_TOT / 128);         // (2, 256)

  // -------- one-time fp32 -> bf16 conversions --------
  {
    int n4;
    n4 = (BN_TOT * P_DIM) / 4;
    cvt_bf16_kernel<<<1024, blk, 0, stream>>>(patches, pb, n4);
    n4 = (P_DIM * H_DIM) / 4;
    cvt_bf16_kernel<<<512, blk, 0, stream>>>(enc_in_w, wib, n4);
    n4 = (DEPTH * H_DIM * H_DIM) / 4;
    cvt_bf16_kernel<<<1024, blk, 0, stream>>>(blk_w1, w1b, n4);
    cvt_bf16_kernel<<<1024, blk, 0, stream>>>(blk_w2, w2b, n4);
    n4 = (H_DIM * NOUT) / 4;
    cvt_bf16_kernel<<<256, blk, 0, stream>>>(enc_out_w, wob, n4);
  }

  // h = gelu(patches @ enc_in_w + b)   (fp32 residual stream)
  wmma_gemm_kernel<1, true, false><<<gH, blk, 0, stream>>>(
      pb, wib, enc_in_b, nullptr, h, nullptr, P_DIM, H_DIM);

  for (int i = 0; i < DEPTH; ++i) {
    ln_kernel<<<BN_TOT, blk, 0, stream>>>(h, blk_ln_g + i * H_DIM,
                                          blk_ln_b + i * H_DIM, xb);
    // t = gelu(x @ w1 + b1)  (bf16 only; feeds next GEMM's A)
    wmma_gemm_kernel<1, false, true><<<gH, blk, 0, stream>>>(
        xb, w1b + (size_t)i * H_DIM * H_DIM, blk_b1 + i * H_DIM,
        nullptr, nullptr, tb, H_DIM, H_DIM);
    // h = h + (t @ w2 + b2)  (fp32 residual + bf16 copy for final GEMM)
    wmma_gemm_kernel<2, true, true><<<gH, blk, 0, stream>>>(
        tb, w2b + (size_t)i * H_DIM * H_DIM, blk_b2 + i * H_DIM,
        h, h, hb, H_DIM, H_DIM);
  }

  // M = h @ enc_out_w + b
  wmma_gemm_kernel<0, true, false><<<gO, blk, 0, stream>>>(
      hb, wob, enc_out_b, nullptr, Mb, nullptr, H_DIM, NOUT);

  finalize_kernel<<<BN_TOT / 256, blk, 0, stream>>>(Mb, (float*)d_out);
}
